// FocusedLinearAttention_14800457302705
// MI455X (gfx1250) — matrix operand order repaired and checked by reading, not compiled, not run on hardware
//
#include <hip/hip_runtime.h>
#include <hip/hip_bf16.h>
#include <stdint.h>

typedef __attribute__((ext_vector_type(16))) __bf16 v16bf;
typedef __attribute__((ext_vector_type(8)))  float  v8f;

#define NHEADS 8
#define BATCH  8
#define SEQ    3136
#define CDIM   512
#define HDIM   64
#define BH     (BATCH*NHEADS)   /* 64  */
#define MTOK   (BATCH*SEQ)      /* 25088 */
#define IMG    56

__device__ inline unsigned short f2bf(float f){
  union { float f; unsigned u; } v; v.f = f;
  unsigned r = v.u + 0x7FFFu + ((v.u >> 16) & 1u);
  return (unsigned short)(r >> 16);
}
__device__ inline float bf2f(unsigned short h){
  union { unsigned u; float f; } v; v.u = ((unsigned)h) << 16; return v.f;
}

union AF { v16bf v; unsigned short u[16]; };

// A fragment (16x32 MxK, bf16): lanes 0-15 hold K 0-7 & 16-23, lanes 16-31 hold +8
__device__ inline int a_kidx(int lane, int i){
  int vv = i >> 1, e = i & 1;
  int k  = (vv < 4) ? (2*vv + e) : (16 + 2*(vv-4) + e);
  return k + ((lane >> 4) << 3);
}
// B fragment (32x16 KxN, bf16): same K order as A (lanes 0-15 hold K 0-7 & 16-23, lanes 16-31 hold +8)
__device__ inline int b_kidx(int lane, int i){
  return a_kidx(lane, i);
}

// ---- CDNA5 async global->LDS DMA (ASYNCcnt path, no VGPR round-trip) -------
__device__ inline void async_g2l_b128(void* lds_dst, const void* gaddr){
  unsigned lds_off = (unsigned)(uintptr_t)lds_dst;
  asm volatile("global_load_async_to_lds_b128 %0, %1, off"
               :: "v"(lds_off), "v"(gaddr) : "memory");
}
template<int N>
__device__ inline void wait_async(){
  asm volatile("s_wait_asynccnt %0" :: "n"(N) : "memory");
}

// ---------------------------------------------------------------- converts
__global__ void k_cvt(const float* __restrict__ src, unsigned short* __restrict__ dst, long n){
  long i = (long)blockIdx.x * blockDim.x + threadIdx.x;
  if (i < n) dst[i] = f2bf(src[i]);
}
__global__ void k_cvt_w(const float* __restrict__ Wq, const float* __restrict__ Wkv,
                        unsigned short* __restrict__ Wb){
  long i = (long)blockIdx.x * blockDim.x + threadIdx.x;
  if (i >= 1536L*512) return;
  int row = (int)(i / 512), col = (int)(i % 512);
  float v = (row < 512) ? Wq[(size_t)row*512 + col] : Wkv[(size_t)(row-512)*512 + col];
  Wb[i] = f2bf(v);
}

// ------------------------------------------------- GEMM1: qkv = x * [Wq;Wkv]^T
// M=25088, K=512, Ncols=1536. Block: 256 thr = 8 waves, tile 128(M) x 64(N).
// Each wave owns a 32x32 macro-tile = 2x2 WMMA accumulators (4 wmma / k-step).
// Double-buffered async DMA: buffer i+1 fills while WMMAs consume buffer i.
__global__ __launch_bounds__(256) void k_gemm_qkv(
    const unsigned short* __restrict__ xb, const unsigned short* __restrict__ Wb,
    const float* __restrict__ pos_enc,
    float* __restrict__ qraw, float* __restrict__ kraw,
    float* __restrict__ v32, unsigned short* __restrict__ vbf)
{
  __shared__ __align__(16) unsigned short sA[2][128][40];
  __shared__ __align__(16) unsigned short sB[2][64][40];
  const int tid  = threadIdx.x;
  const int lane = tid & 31;
  const int w    = tid >> 5;
  const int m0   = blockIdx.y * 128;
  const int n0   = blockIdx.x * 64;
  const int wm   = w >> 1, wn = w & 1;
  v8f acc[2][2] = {};

  auto load_tile = [&](int buf, int k0){     // 3 async ops per thread/wave
    #pragma unroll
    for (int it = 0; it < 2; it++) {
      int c = tid + it*256;
      int row = c >> 2, chh = c & 3;
      async_g2l_b128(&sA[buf][row][chh*8], xb + (size_t)(m0+row)*CDIM + k0 + chh*8);
    }
    int row = tid >> 2, chh = tid & 3;
    async_g2l_b128(&sB[buf][row][chh*8], Wb + (size_t)(n0+row)*CDIM + k0 + chh*8);
  };
  auto compute = [&](int buf){
    AF a[2], b[2];
    int ml = lane & 15;
    #pragma unroll
    for (int mt = 0; mt < 2; mt++)
      #pragma unroll
      for (int i = 0; i < 16; i++) a[mt].u[i] = sA[buf][wm*32 + mt*16 + ml][a_kidx(lane, i)];
    #pragma unroll
    for (int nt = 0; nt < 2; nt++)
      #pragma unroll
      for (int i = 0; i < 16; i++) b[nt].u[i] = sB[buf][wn*32 + nt*16 + ml][b_kidx(lane, i)];
    #pragma unroll
    for (int mt = 0; mt < 2; mt++)
      #pragma unroll
      for (int nt = 0; nt < 2; nt++)
        acc[mt][nt] = __builtin_amdgcn_wmma_f32_16x16x32_bf16(
            false, a[mt].v, false, b[nt].v, (short)0, acc[mt][nt], false, false);
  };

  load_tile(0, 0);
  for (int k0 = 0; k0 < CDIM; k0 += 64) {    // 8 iterations, 2 k-steps each
    load_tile(1, k0 + 32);                   // k0+32 <= 480, always valid
    wait_async<3>();                         // older batch (buf0) complete
    __syncthreads();
    compute(0);
    __syncthreads();
    if (k0 + 64 < CDIM) { load_tile(0, k0 + 64); wait_async<3>(); }
    else                  wait_async<0>();
    __syncthreads();
    compute(1);
    __syncthreads();
  }

  #pragma unroll
  for (int mt = 0; mt < 2; mt++) {
    int mbase = m0 + wm*32 + mt*16 + ((lane >> 4) << 3);
    #pragma unroll
    for (int nt = 0; nt < 2; nt++) {
      int ncol = n0 + wn*32 + nt*16 + (lane & 15);
      #pragma unroll
      for (int r = 0; r < 8; r++) {
        int token = mbase + r;
        int bidx = token / SEQ, nseq = token % SEQ;
        float val = acc[mt][nt][r];
        if (ncol < 512) {                     // q
          int head = ncol >> 6, ch = ncol & 63;
          qraw[(((size_t)(bidx*NHEADS+head))*SEQ + nseq)*HDIM + ch] = val;
        } else if (ncol < 1024) {             // k (+pos_enc)
          int c2 = ncol - 512;
          val += pos_enc[(size_t)nseq*CDIM + c2];
          int head = c2 >> 6, ch = c2 & 63;
          kraw[(((size_t)(bidx*NHEADS+head))*SEQ + nseq)*HDIM + ch] = val;
        } else {                              // v (f32 for conv + bf16 for k^T v)
          int c2 = ncol - 1024;
          int head = c2 >> 6, ch = c2 & 63;
          size_t o = (((size_t)(bidx*NHEADS+head))*SEQ + nseq)*HDIM + ch;
          v32[o] = val; vbf[o] = f2bf(val);
        }
      }
    }
  }
}

// ------------------------------------------- _focus: one wave per 64-elem row
__global__ __launch_bounds__(256) void k_focus(
    const float* __restrict__ qraw, const float* __restrict__ kraw,
    unsigned short* __restrict__ qf, unsigned short* __restrict__ kf)
{
  __shared__ float sv[8][64];
  __shared__ float sw[8][64];
  const int tid = threadIdx.x, lane = tid & 31, w = tid >> 5;
  long row = (long)blockIdx.x * 8 + w;
  const long NROW = (long)BH * SEQ;
  const float* src; unsigned short* dst; long r2;
  if (row < NROW) { src = qraw; dst = qf; r2 = row; }
  else            { src = kraw; dst = kf; r2 = row - NROW; }
  const float* p = src + r2 * 64;
  float a0 = fmaxf(p[lane],      0.f) + 1e-6f;
  float a1 = fmaxf(p[lane + 32], 0.f) + 1e-6f;
  float ss = a0*a0 + a1*a1;
  #pragma unroll
  for (int m = 16; m >= 1; m >>= 1) ss += __shfl_xor(ss, m, 32);
  float nrm = sqrtf(ss);
  sv[w][lane] = a0; sv[w][lane + 32] = a1;
  __syncthreads();
  #pragma unroll
  for (int jj = 0; jj < 2; jj++) {        // stable argsort via rank counting
    int j = lane + jj*32;
    float val = sv[w][j];
    int rank = 0;
    for (int t = 0; t < 64; t++) {
      float vt = sv[w][t];
      rank += (vt < val) || (vt == val && t < j);
    }
    sw[w][rank] = (float)j;               // idx[p] = index of p-th smallest
  }
  __syncthreads();
  float b0 = a0 * sw[w][lane];
  float b1 = a1 * sw[w][lane + 32];
  float s2 = b0*b0 + b1*b1;
  #pragma unroll
  for (int m = 16; m >= 1; m >>= 1) s2 += __shfl_xor(s2, m, 32);
  float scale = nrm / sqrtf(s2);
  dst[r2*64 + lane]      = f2bf(b0 * scale);
  dst[r2*64 + lane + 32] = f2bf(b1 * scale);
}

// -------------------------------------------------- ksum[bh][c] = sum_n k[n,c]
__global__ void k_ksum(const unsigned short* __restrict__ kf, float* __restrict__ ksum){
  int bh = blockIdx.x, ch = threadIdx.x;   // 64 threads
  float acc = 0.f;
  const unsigned short* p = kf + (size_t)bh*SEQ*64 + ch;
  for (int n = 0; n < SEQ; n++) acc += bf2f(p[(size_t)n*64]);
  ksum[bh*64 + ch] = acc;
}

// ------------------------------ kv_ctx[bh] = k^T v  (64x64, K=3136), 16 waves
// Double-buffered async DMA: 64 sequence rows per phase, 2 wmma per phase.
__global__ __launch_bounds__(512) void k_kvctx(
    const unsigned short* __restrict__ kf, const unsigned short* __restrict__ vbf,
    unsigned short* __restrict__ kvbf)
{
  __shared__ __align__(16) unsigned short sK[2][64][72];
  __shared__ __align__(16) unsigned short sV[2][64][72];
  int tid = threadIdx.x, lane = tid & 31, w = tid >> 5;
  int bh = blockIdx.x;
  int mi = w >> 2, ni = w & 3;
  v8f acc = {};
  const unsigned short* kb = kf  + (size_t)bh*SEQ*64;
  const unsigned short* vb = vbf + (size_t)bh*SEQ*64;

  auto load_tile = [&](int buf, int n0){    // 2 async ops per thread/wave
    #pragma unroll
    for (int it = 0; it < 2; it++) {
      int c = (tid & 255) + it*256;
      int rowl = c >> 3, ch8 = (c & 7) * 8;
      if (tid < 256) async_g2l_b128(&sK[buf][rowl][ch8], kb + (size_t)(n0+rowl)*64 + ch8);
      else           async_g2l_b128(&sV[buf][rowl][ch8], vb + (size_t)(n0+rowl)*64 + ch8);
    }
  };
  auto compute = [&](int buf){
    int m = mi*16 + (lane & 15);
    int n = ni*16 + (lane & 15);
    #pragma unroll
    for (int kk0 = 0; kk0 < 64; kk0 += 32) {
      AF a, b;
      #pragma unroll
      for (int i = 0; i < 16; i++) a.u[i] = sK[buf][kk0 + a_kidx(lane, i)][m]; // transposed
      #pragma unroll
      for (int i = 0; i < 16; i++) b.u[i] = sV[buf][kk0 + b_kidx(lane, i)][n];
      acc = __builtin_amdgcn_wmma_f32_16x16x32_bf16(false, a.v, false, b.v,
                                                    (short)0, acc, false, false);
    }
  };

  load_tile(0, 0);
  int n0 = 0;
  for (; n0 + 64 < SEQ; n0 += 128) {        // pairs: n0 -> buf0, n0+64 -> buf1
    load_tile(1, n0 + 64);
    wait_async<2>();
    __syncthreads();
    compute(0);
    __syncthreads();
    if (n0 + 128 < SEQ) { load_tile(0, n0 + 128); wait_async<2>(); }
    else                  wait_async<0>();
    __syncthreads();
    compute(1);
    __syncthreads();
  }
  if (n0 < SEQ) {                           // SEQ/64 = 49 steps -> odd tail
    wait_async<0>();
    __syncthreads();
    compute(0);
    __syncthreads();
  }

  int ncol  = ni*16 + (lane & 15);
  int mrow0 = mi*16 + ((lane >> 4) << 3);
  #pragma unroll
  for (int r = 0; r < 8; r++)
    kvbf[(size_t)bh*4096 + (size_t)(mrow0 + r)*64 + ncol] = f2bf(acc[r]);
}

// --------------------- out[bh,i,:] = z_i * q[i,:] @ kv_ctx ; z = 1/(q.ksum+eps)
__global__ __launch_bounds__(128) void k_attn(
    const unsigned short* __restrict__ qf, const unsigned short* __restrict__ kvbf,
    const float* __restrict__ ksum, float* __restrict__ out32)
{
  __shared__ __align__(16) unsigned short sQ[16][72];
  __shared__ float sZ[16];
  int tid = threadIdx.x, lane = tid & 31, w = tid >> 5;  // 4 waves
  int bh = blockIdx.y;
  int m0 = blockIdx.x * 16;
  const unsigned short* qb = qf + ((size_t)bh*SEQ + m0) * 64;
  { int rowl = tid >> 3, ch8 = (tid & 7) * 8;
    *(uint4*)&sQ[rowl][ch8] = *(const uint4*)(qb + (size_t)rowl*64 + ch8); }
  __syncthreads();
  if (tid < 16) {
    float d = 0.f;
    const float* ks = ksum + bh*64;
    for (int c = 0; c < 64; c++) d += bf2f(sQ[tid][c]) * ks[c];
    sZ[tid] = 1.0f / (d + 1e-6f);
  }
  v8f acc = {};
  const unsigned short* kvb = kvbf + (size_t)bh*4096;
  #pragma unroll
  for (int k0 = 0; k0 < 64; k0 += 32) {
    AF a, b;
    int m = lane & 15;
    int n = w*16 + (lane & 15);
    #pragma unroll
    for (int i = 0; i < 16; i++) a.u[i] = sQ[m][k0 + a_kidx(lane, i)];
    #pragma unroll
    for (int i = 0; i < 16; i++) b.u[i] = kvb[(size_t)(k0 + b_kidx(lane, i))*64 + n];
    acc = __builtin_amdgcn_wmma_f32_16x16x32_bf16(false, a.v, false, b.v,
                                                  (short)0, acc, false, false);
  }
  __syncthreads();
  int ncol = w*16 + (lane & 15);
  int mr0  = (lane >> 4) << 3;
  #pragma unroll
  for (int r = 0; r < 8; r++) {
    int mr = mr0 + r;
    out32[((size_t)bh*SEQ + m0 + mr)*64 + ncol] = acc[r] * sZ[mr];
  }
}

// -------------------- depthwise 5x5 conv on v + residual add + head merge -> bf16
__global__ void k_conv_merge(
    const float* __restrict__ v32, const float* __restrict__ attn32,
    const float* __restrict__ dwc_w, const float* __restrict__ dwc_b,
    unsigned short* __restrict__ merged)
{
  long idx = (long)blockIdx.x * blockDim.x + threadIdx.x;
  long total = (long)BH * SEQ * 64;
  if (idx >= total) return;
  int ch = idx & 63;
  long t = idx >> 6;
  int n  = (int)(t % SEQ);
  int bh = (int)(t / SEQ);
  int y = n / IMG, x = n % IMG;
  float acc = dwc_b[ch];
  const float* vb = v32 + (size_t)bh*SEQ*64;
  #pragma unroll
  for (int ky = 0; ky < 5; ky++) {
    int yy = y + ky - 2;
    if (yy < 0 || yy >= IMG) continue;
    #pragma unroll
    for (int kx = 0; kx < 5; kx++) {
      int xx = x + kx - 2;
      if (xx < 0 || xx >= IMG) continue;
      acc += dwc_w[ch*25 + ky*5 + kx] * vb[((size_t)(yy*IMG + xx))*64 + ch];
    }
  }
  float o = attn32[idx] + acc;
  int b = bh >> 3, head = bh & 7;
  merged[((size_t)(b*SEQ + n))*CDIM + head*64 + ch] = f2bf(o);
}

// ------------------------------------ GEMM3: out = merged * Wproj^T + bproj (f32)
__global__ __launch_bounds__(256) void k_gemm_proj(
    const unsigned short* __restrict__ Ab, const unsigned short* __restrict__ Wpb,
    const float* __restrict__ bias, float* __restrict__ out)
{
  __shared__ __align__(16) unsigned short sA[2][128][40];
  __shared__ __align__(16) unsigned short sB[2][64][40];
  const int tid  = threadIdx.x;
  const int lane = tid & 31;
  const int w    = tid >> 5;
  const int m0   = blockIdx.y * 128;
  const int n0   = blockIdx.x * 64;
  const int wm   = w >> 1, wn = w & 1;
  v8f acc[2][2] = {};

  auto load_tile = [&](int buf, int k0){
    #pragma unroll
    for (int it = 0; it < 2; it++) {
      int c = tid + it*256;
      int row = c >> 2, chh = c & 3;
      async_g2l_b128(&sA[buf][row][chh*8], Ab + (size_t)(m0+row)*CDIM + k0 + chh*8);
    }
    int row = tid >> 2, chh = tid & 3;
    async_g2l_b128(&sB[buf][row][chh*8], Wpb + (size_t)(n0+row)*CDIM + k0 + chh*8);
  };
  auto compute = [&](int buf){
    AF a[2], b[2];
    int ml = lane & 15;
    #pragma unroll
    for (int mt = 0; mt < 2; mt++)
      #pragma unroll
      for (int i = 0; i < 16; i++) a[mt].u[i] = sA[buf][wm*32 + mt*16 + ml][a_kidx(lane, i)];
    #pragma unroll
    for (int nt = 0; nt < 2; nt++)
      #pragma unroll
      for (int i = 0; i < 16; i++) b[nt].u[i] = sB[buf][wn*32 + nt*16 + ml][b_kidx(lane, i)];
    #pragma unroll
    for (int mt = 0; mt < 2; mt++)
      #pragma unroll
      for (int nt = 0; nt < 2; nt++)
        acc[mt][nt] = __builtin_amdgcn_wmma_f32_16x16x32_bf16(
            false, a[mt].v, false, b[nt].v, (short)0, acc[mt][nt], false, false);
  };

  load_tile(0, 0);
  for (int k0 = 0; k0 < CDIM; k0 += 64) {
    load_tile(1, k0 + 32);
    wait_async<3>();
    __syncthreads();
    compute(0);
    __syncthreads();
    if (k0 + 64 < CDIM) { load_tile(0, k0 + 64); wait_async<3>(); }
    else                  wait_async<0>();
    __syncthreads();
    compute(1);
    __syncthreads();
  }

  #pragma unroll
  for (int mt = 0; mt < 2; mt++) {
    int mbase = m0 + wm*32 + mt*16 + ((lane >> 4) << 3);
    #pragma unroll
    for (int nt = 0; nt < 2; nt++) {
      int ncol = n0 + wn*32 + nt*16 + (lane & 15);
      #pragma unroll
      for (int r = 0; r < 8; r++) {
        int token = mbase + r;
        out[(size_t)token*CDIM + ncol] = acc[mt][nt][r] + bias[ncol];
      }
    }
  }
}

// ---------------------------------------------------------------------- launch
extern "C" void kernel_launch(void* const* d_in, const int* in_sizes, int n_in,
                              void* d_out, int out_size, void* d_ws, size_t ws_size,
                              hipStream_t stream)
{
  (void)in_sizes; (void)n_in; (void)out_size; (void)ws_size;
  const float* x      = (const float*)d_in[0];
  const float* Wq     = (const float*)d_in[1];
  const float* Wkv    = (const float*)d_in[2];
  const float* pos    = (const float*)d_in[3];
  const float* dwc_w  = (const float*)d_in[4];
  const float* dwc_b  = (const float*)d_in[5];
  const float* Wproj  = (const float*)d_in[6];
  const float* bproj  = (const float*)d_in[7];
  float* out = (float*)d_out;

  char* ws = (char*)d_ws;
  size_t off = 0;
  auto alloc = [&](size_t bytes) -> char* {
    char* p = ws + off;
    off = (off + bytes + 255) & ~(size_t)255;
    return p;
  };
  const size_t NE  = (size_t)BH * SEQ * HDIM;   // 12.85M elems per head-tensor
  unsigned short* xb    = (unsigned short*)alloc((size_t)MTOK*CDIM*2);
  unsigned short* Wb    = (unsigned short*)alloc(1536UL*512*2);
  unsigned short* Wpb   = (unsigned short*)alloc(512UL*512*2);
  float*          qraw  = (float*)         alloc(NE*4);
  float*          kraw  = (float*)         alloc(NE*4);
  float*          v32   = (float*)         alloc(NE*4);
  unsigned short* vbf   = (unsigned short*)alloc(NE*2);
  unsigned short* qf    = (unsigned short*)alloc(NE*2);
  unsigned short* kf    = (unsigned short*)alloc(NE*2);
  float*          ksum  = (float*)         alloc((size_t)BH*64*4);
  unsigned short* kvbf  = (unsigned short*)alloc((size_t)BH*4096*2);
  // alias: qraw dead after k_focus -> reuse as attn32; kraw dead -> reuse as merged
  float*          attn32 = qraw;
  unsigned short* merged = (unsigned short*)kraw;

  long nx = (long)MTOK * CDIM;
  k_cvt   <<<(nx + 255)/256, 256, 0, stream>>>(x, xb, nx);
  k_cvt_w <<<(1536L*512 + 255)/256, 256, 0, stream>>>(Wq, Wkv, Wb);
  long nwp = 512L*512;
  k_cvt   <<<(nwp + 255)/256, 256, 0, stream>>>(Wproj, Wpb, nwp);

  k_gemm_qkv<<<dim3(24, 196), 256, 0, stream>>>(xb, Wb, pos, qraw, kraw, v32, vbf);
  k_focus   <<<50176, 256, 0, stream>>>(qraw, kraw, qf, kf);
  k_ksum    <<<64, 64, 0, stream>>>(kf, ksum);
  k_kvctx   <<<64, 512, 0, stream>>>(kf, vbf, kvbf);
  k_attn    <<<dim3(196, 64), 128, 0, stream>>>(qf, kvbf, ksum, attn32);

  long nc = (long)NE;
  k_conv_merge<<<(nc + 255)/256, 256, 0, stream>>>(v32, attn32, dwc_w, dwc_b, merged);
  k_gemm_proj <<<dim3(8, 196), 256, 0, stream>>>(merged, Wpb, bproj, out);
}
